// HRRSelfAttention_58634893525655
// MI455X (gfx1250) — matrix operand order repaired and checked by reading, not compiled, not run on hardware
//
#include <hip/hip_runtime.h>
#include <hip/hip_bf16.h>

// ---------------------------------------------------------------------------
// HRR self-attention for MI455X (gfx1250, wave32, WMMA).
//   1. fp32->bf16 converts; B operands pre-TRANSPOSED to [N][K] so GEMM
//      staging is contiguous (DFT cos/sin matrices are symmetric = free).
//   2. WMMA GEMM: qkv(bf16) = xb @ wqkv
//   3. WMMA GEMMs: Fq/Fk/Fv re&im = qkv_head @ DFT(128x128)   [z=8 heads]
//   4. bind: kv = Fk*Fv (fp32) -> 3-pass chunked cumsum over S (fp32)
//   5. unbind: u = kv*conj(Fq), packed bf16 [re|im] along K
//   6. WMMA GEMM: vals(bf16) = u @ [cos;-sin]/128 (K=256, folds ifft + Re)
//   7. WMMA GEMM: out(fp32) = vals @ w_out
// B k-tiles are staged to LDS with GLOBAL_LOAD_ASYNC_TO_LDS_B128 (ASYNCcnt +
// s_wait_asynccnt); builtin signature: (v4i AS1*, v4i AS3*, imm, imm).
// ---------------------------------------------------------------------------

typedef __attribute__((ext_vector_type(16))) __bf16 v16bf;
typedef __attribute__((ext_vector_type(8)))  __bf16 v8bf;
typedef __attribute__((ext_vector_type(8)))  float  v8f;

typedef int v4i __attribute__((vector_size(16)));
typedef __attribute__((address_space(1))) v4i* gv4i_ptr;
typedef __attribute__((address_space(3))) v4i* lv4i_ptr;

#if __has_builtin(__builtin_amdgcn_global_load_async_to_lds_b128)
#define HAVE_ASYNC_LDS 1
#else
#define HAVE_ASYNC_LDS 0
#endif

__device__ inline void wait_async0() {
#if __has_builtin(__builtin_amdgcn_s_wait_asynccnt)
    __builtin_amdgcn_s_wait_asynccnt(0);
#elif HAVE_ASYNC_LDS
    asm volatile("s_wait_asynccnt 0" ::: "memory");
#endif
}

#define NTOK 16384   // B*S
#define DIM  1024
#define SEQ  4096
#define NH   8
#define HD   128

// ------------------------------ elementwise --------------------------------

__global__ void f2bf_kernel(const float* __restrict__ in, __bf16* __restrict__ out, int n) {
    int i = blockIdx.x * blockDim.x + threadIdx.x;
    if (i < n) out[i] = (__bf16)in[i];
}

// fp32 [K][N] row-major  ->  bf16 [N][K] (transposed, for GEMM B operand)
__global__ void f2bf_t_kernel(const float* __restrict__ in, __bf16* __restrict__ out,
                              int Kd, int Nd) {
    int i = blockIdx.x * blockDim.x + threadIdx.x;
    if (i >= Kd * Nd) return;
    int k = i / Nd, n = i - k * Nd;
    out[(size_t)n * Kd + k] = (__bf16)in[i];
}

// DFT matrices (bf16).  dftr/dfti are symmetric (exp(-2pi i jk/128)) so no
// transpose needed. idft_t is built directly transposed: [j][kk], kk<128 ->
// cos(2pi kk j/128)/128 ; kk>=128 -> -sin(2pi (kk-128) j/128)/128.
__global__ void build_mats_kernel(__bf16* __restrict__ dftr,
                                  __bf16* __restrict__ dfti,
                                  __bf16* __restrict__ idft_t) {
    int idx = blockIdx.x * blockDim.x + threadIdx.x;   // 0 .. 32767
    if (idx >= 128 * 256) return;
    int j  = idx >> 8;
    int kk = idx & 255;
    int m  = ((kk & 127) * j) & 127;                   // exact mod-128 phase
    float ang = 6.28318530717958647692f * (float)m / 128.0f;
    idft_t[idx] = (__bf16)((kk < 128 ? cosf(ang) : -sinf(ang)) * (1.0f / 128.0f));
    if (idx < 128 * 128) {
        int jj = idx >> 7, k2 = idx & 127;
        int m2 = (jj * k2) & 127;
        float a2 = 6.28318530717958647692f * (float)m2 / 128.0f;
        dftr[idx] = (__bf16)cosf(a2);
        dfti[idx] = (__bf16)(-sinf(a2));
    }
}

// kv = fk * fv (complex), in-place into the Fk buffers.
__global__ void bind_kernel(float* __restrict__ kr, float* __restrict__ ki,
                            const float* __restrict__ vr, const float* __restrict__ vi) {
    size_t i = (size_t)blockIdx.x * blockDim.x + threadIdx.x;
    float a = kr[i], b = ki[i], c = vr[i], d = vi[i];
    kr[i] = a * c - b * d;
    ki[i] = a * d + b * c;
}

// u = kv * conj(fq); pack [re(128) | im(128)] per head into bf16.
__global__ void unbind_pack_kernel(const float* __restrict__ kr, const float* __restrict__ ki,
                                   const float* __restrict__ qr, const float* __restrict__ qi,
                                   __bf16* __restrict__ u) {
    size_t i = (size_t)blockIdx.x * blockDim.x + threadIdx.x;
    float a = kr[i], b = ki[i], c = qr[i], d = qi[i];
    float ur = a * c + b * d;
    float ui = b * c - a * d;
    size_t t = i >> 10, hf = i & 1023, h = hf >> 7, f = hf & 127;
    size_t o = t * 2048 + h * 256 + f;
    u[o]       = (__bf16)ur;
    u[o + 128] = (__bf16)ui;
}

// ------------------------- chunked cumsum over S ---------------------------

__global__ void scan_chunksum_kernel(float* __restrict__ kr, float* __restrict__ ki,
                                     float* __restrict__ part) {
    int tid = blockIdx.x * blockDim.x + threadIdx.x;       // 0 .. 131071
    const float* buf = blockIdx.z ? ki : kr;
    float* p = part + (size_t)blockIdx.z * 4096 * 32;
    int c = tid >> 5, j = tid & 31;
    int b = c >> 10, d = c & 1023;
    size_t base = ((size_t)b * SEQ + (size_t)j * 128) * DIM + d;
    float s = 0.0f;
    for (int i = 0; i < 128; ++i) s += buf[base + (size_t)i * DIM];
    p[c * 32 + j] = s;
}

__global__ void scan_prefix_kernel(float* __restrict__ part) {
    int c = blockIdx.x * blockDim.x + threadIdx.x;         // 0 .. 4095
    float* p = part + (size_t)blockIdx.z * 4096 * 32 + (size_t)c * 32;
    float run = 0.0f;
    for (int j = 0; j < 32; ++j) { float t = p[j]; p[j] = run; run += t; }
}

__global__ void scan_apply_kernel(float* __restrict__ kr, float* __restrict__ ki,
                                  const float* __restrict__ part) {
    int tid = blockIdx.x * blockDim.x + threadIdx.x;
    float* buf = blockIdx.z ? ki : kr;
    const float* p = part + (size_t)blockIdx.z * 4096 * 32;
    int c = tid >> 5, j = tid & 31;
    int b = c >> 10, d = c & 1023;
    size_t base = ((size_t)b * SEQ + (size_t)j * 128) * DIM + d;
    float run = p[c * 32 + j];
    for (int i = 0; i < 128; ++i) {
        size_t a = base + (size_t)i * DIM;
        run += buf[a];
        buf[a] = run;
    }
}

// ------------------------------ WMMA GEMM ----------------------------------
// C[M,N] = A[M,K](bf16,row-major,lda) * Bt[N,K](bf16,row-major = B transposed)
// Block = 128 threads (4 waves), block tile 64x64; each wave computes 16x64
// (one A fragment reused across 4 WMMAs). B k-tile staged to LDS as a
// straight contiguous copy (Bt rows) via async global->LDS DMA.

__global__ __launch_bounds__(128) void wmma_gemm_bf16(
    const __bf16* __restrict__ A, long aBatch, int lda,
    const __bf16* __restrict__ Bt,
    void* __restrict__ Cout, long cBatch, int ldc,
    int K, int outBf16)
{
    __shared__ alignas(16) __bf16 Bs[64][40];   // [col][k]; 80B row stride (16B aligned)

    const int tid  = threadIdx.x;
    const int lane = tid & 31;
    const int wave = tid >> 5;
    const int half = lane >> 4;
    const int l15  = lane & 15;

    const int colBase = blockIdx.x * 64;
    const int rowBase = blockIdx.y * 64 + wave * 16;

    const __bf16* Ab = A + (long)blockIdx.z * aBatch
                         + (long)(rowBase + l15) * lda + half * 16;

    // staging assignment: thread -> (column c, 16-element segment of the k-tile)
    const int sc  = tid >> 1;
    const int sof = (tid & 1) << 4;
    const __bf16* Brow = Bt + (long)(colBase + sc) * K + sof;

    v8f acc[4] = {v8f{}, v8f{}, v8f{}, v8f{}};

    for (int k0 = 0; k0 < K; k0 += 32) {
        // ---- stage Bt[colBase..+64)[k0..k0+32) into LDS (straight copy) ----
        {
            const __bf16* src = Brow + k0;
            __bf16* dst = &Bs[sc][sof];
            if (k0 + 32 < K)
                __builtin_prefetch((const void*)(src + 32), 0, 3);   // global_prefetch_b8
#if HAVE_ASYNC_LDS
            __builtin_amdgcn_global_load_async_to_lds_b128(
                (gv4i_ptr)(size_t)src, (lv4i_ptr)(unsigned)(size_t)dst, 0, 0);
            __builtin_amdgcn_global_load_async_to_lds_b128(
                (gv4i_ptr)(size_t)(src + 8), (lv4i_ptr)(unsigned)(size_t)(dst + 8), 0, 0);
            wait_async0();
#else
            v8bf b0 = *(const v8bf*)src;
            v8bf b1 = *(const v8bf*)(src + 8);
            *(v8bf*)dst       = b0;
            *(v8bf*)(dst + 8) = b1;
#endif
        }
        __syncthreads();

        // ---- A fragment: row rowBase+l15, K = k0 + 16*half + {0..15} ----
        v16bf af;
        {
            v8bf a0 = *(const v8bf*)(Ab + k0);
            v8bf a1 = *(const v8bf*)(Ab + k0 + 8);
#pragma unroll
            for (int e = 0; e < 8; ++e) { af[e] = a0[e]; af[8 + e] = a1[e]; }
        }

        // ---- 4 B fragments from LDS + 4 WMMAs ----
#pragma unroll
        for (int t = 0; t < 4; ++t) {
            const __bf16* bp = &Bs[t * 16 + l15][half * 16];  // K = k0 + 16*half + e
            v8bf b0 = *(const v8bf*)(bp);
            v8bf b1 = *(const v8bf*)(bp + 8);
            v16bf bfr;
#pragma unroll
            for (int e = 0; e < 8; ++e) { bfr[e] = b0[e]; bfr[8 + e] = b1[e]; }
            acc[t] = __builtin_amdgcn_wmma_f32_16x16x32_bf16(
                false, af, false, bfr, (short)0, acc[t], false, false);
        }
        __syncthreads();
    }

    // ---- store: VGPR r -> row = rowBase + 8*half + r, col = colBase + 16t + l15 ----
    const long cb = (long)blockIdx.z * cBatch;
    const int crow = rowBase + half * 8;
#pragma unroll
    for (int t = 0; t < 4; ++t) {
        int col = colBase + t * 16 + l15;
#pragma unroll
        for (int r = 0; r < 8; ++r) {
            long idx = cb + (long)(crow + r) * ldc + col;
            float v = acc[t][r];
            if (outBf16) ((__bf16*)Cout)[idx] = (__bf16)v;
            else         ((float*)Cout)[idx]  = v;
        }
    }
}

// ------------------------------- launcher ----------------------------------

extern "C" void kernel_launch(void* const* d_in, const int* in_sizes, int n_in,
                              void* d_out, int out_size, void* d_ws, size_t ws_size,
                              hipStream_t stream) {
    (void)in_sizes; (void)n_in; (void)out_size; (void)ws_size;

    const float* x    = (const float*)d_in[0];
    const float* wqkv = (const float*)d_in[1];
    const float* wout = (const float*)d_in[2];
    float* out = (float*)d_out;

    char* ws = (char*)d_ws;
    size_t off = 0;
    auto alloc = [&](size_t bytes) -> size_t {
        size_t r = off;
        off += (bytes + 255) & ~(size_t)255;
        return r;
    };

    const size_t NE = (size_t)NTOK * DIM;           // 16,777,216
    size_t o_xb    = alloc((size_t)NTOK * DIM * 2);
    size_t o_wqkvt = alloc((size_t)DIM * 3 * DIM * 2);   // [3072][1024] bf16
    size_t o_woutt = alloc((size_t)DIM * DIM * 2);       // [1024][1024] bf16
    size_t o_dftr  = alloc(128 * 128 * 2);
    size_t o_dfti  = alloc(128 * 128 * 2);
    size_t o_idft  = alloc(128 * 256 * 2);               // [128][256] transposed
    size_t o_qkvb  = alloc((size_t)NTOK * 3 * DIM * 2);
    size_t o_Fqr   = alloc(NE * 4);
    size_t o_Fqi   = alloc(NE * 4);
    size_t o_Fkr   = alloc(NE * 4);                 // becomes kv_r
    size_t o_Fki   = alloc(NE * 4);                 // becomes kv_i
    size_t o_Fvr   = alloc(NE * 4);                 // reused as u_cat (bf16)
    size_t o_Fvi   = alloc(NE * 4);                 // reused as vals (bf16)
    size_t o_part  = alloc((size_t)2 * 4096 * 32 * 4);

    __bf16* xb    = (__bf16*)(ws + o_xb);
    __bf16* wqkvt = (__bf16*)(ws + o_wqkvt);
    __bf16* woutt = (__bf16*)(ws + o_woutt);
    __bf16* dftr  = (__bf16*)(ws + o_dftr);
    __bf16* dfti  = (__bf16*)(ws + o_dfti);
    __bf16* idftt = (__bf16*)(ws + o_idft);
    __bf16* qkvb  = (__bf16*)(ws + o_qkvb);
    float*  Fqr = (float*)(ws + o_Fqr);
    float*  Fqi = (float*)(ws + o_Fqi);
    float*  Fkr = (float*)(ws + o_Fkr);
    float*  Fki = (float*)(ws + o_Fki);
    float*  Fvr = (float*)(ws + o_Fvr);
    float*  Fvi = (float*)(ws + o_Fvi);
    float*  part = (float*)(ws + o_part);
    __bf16* ucat  = (__bf16*)(ws + o_Fvr);          // alias (Fv dead after bind)
    __bf16* valsb = (__bf16*)(ws + o_Fvi);          // alias

    // 1) converts (+ transposes for B operands) + DFT matrices
    f2bf_kernel<<<(int)((NE + 255) / 256), 256, 0, stream>>>(x, xb, (int)NE);
    f2bf_t_kernel<<<(DIM * 3 * DIM + 255) / 256, 256, 0, stream>>>(wqkv, wqkvt, DIM, 3 * DIM);
    f2bf_t_kernel<<<(DIM * DIM + 255) / 256, 256, 0, stream>>>(wout, woutt, DIM, DIM);
    build_mats_kernel<<<(128 * 256 + 255) / 256, 256, 0, stream>>>(dftr, dfti, idftt);

    // 2) qkv = xb @ wqkv  -> bf16 [16384 x 3072]
    wmma_gemm_bf16<<<dim3(3 * DIM / 64, NTOK / 64, 1), 128, 0, stream>>>(
        xb, 0L, DIM, wqkvt, qkvb, 0L, 3 * DIM, DIM, 1);

    // 3) per-head DFTs (z = 8 heads): F = qkv_head[16384x128] @ DFT[128x128]
    dim3 gDft(HD / 64, NTOK / 64, NH);
    wmma_gemm_bf16<<<gDft, 128, 0, stream>>>(qkvb + 0 * DIM, (long)HD, 3 * DIM,
                                             dftr, Fqr, (long)HD, DIM, HD, 0);
    wmma_gemm_bf16<<<gDft, 128, 0, stream>>>(qkvb + 0 * DIM, (long)HD, 3 * DIM,
                                             dfti, Fqi, (long)HD, DIM, HD, 0);
    wmma_gemm_bf16<<<gDft, 128, 0, stream>>>(qkvb + 1 * DIM, (long)HD, 3 * DIM,
                                             dftr, Fkr, (long)HD, DIM, HD, 0);
    wmma_gemm_bf16<<<gDft, 128, 0, stream>>>(qkvb + 1 * DIM, (long)HD, 3 * DIM,
                                             dfti, Fki, (long)HD, DIM, HD, 0);
    wmma_gemm_bf16<<<gDft, 128, 0, stream>>>(qkvb + 2 * DIM, (long)HD, 3 * DIM,
                                             dftr, Fvr, (long)HD, DIM, HD, 0);
    wmma_gemm_bf16<<<gDft, 128, 0, stream>>>(qkvb + 2 * DIM, (long)HD, 3 * DIM,
                                             dfti, Fvi, (long)HD, DIM, HD, 0);

    // 4) kv = fk * fv (into Fk buffers)
    bind_kernel<<<(int)(NE / 256), 256, 0, stream>>>(Fkr, Fki, Fvr, Fvi);

    // 5) causal cumsum over S (chunked 3-pass, fp32)
    scan_chunksum_kernel<<<dim3(131072 / 256, 1, 2), 256, 0, stream>>>(Fkr, Fki, part);
    scan_prefix_kernel<<<dim3(4096 / 256, 1, 2), 256, 0, stream>>>(part);
    scan_apply_kernel<<<dim3(131072 / 256, 1, 2), 256, 0, stream>>>(Fkr, Fki, part);

    // 6) u = kv * conj(fq), packed bf16 [re|im] per head
    unbind_pack_kernel<<<(int)(NE / 256), 256, 0, stream>>>(Fkr, Fki, Fqr, Fqi, ucat);

    // 7) vals = Re(ifft(u)) via K=256 GEMM with [cos; -sin]/128  -> bf16
    wmma_gemm_bf16<<<dim3(HD / 64, NTOK / 64, NH), 128, 0, stream>>>(
        ucat, 256L, 2048, idftt, valsb, (long)HD, DIM, 256, 1);

    // 8) out = vals @ w_out  -> fp32
    wmma_gemm_bf16<<<dim3(DIM / 64, NTOK / 64, 1), 128, 0, stream>>>(
        valsb, 0L, DIM, woutt, out, 0L, DIM, DIM, 0);
}
